// SegXAttn_77343771066721
// MI455X (gfx1250) — matrix operand order repaired and checked by reading, not compiled, not run on hardware
//
#include <hip/hip_runtime.h>
#include <hip/hip_bf16.h>

// ---------------- CDNA5 WMMA types ----------------
typedef __attribute__((ext_vector_type(16))) __bf16 v16bf;
typedef __attribute__((ext_vector_type(8)))  float  v8f;

union Frag {
    v16bf v;
    uint4 u[2];
};

__device__ __forceinline__ v8f wmma_bf16(const Frag& a, const Frag& b, v8f c) {
    return __builtin_amdgcn_wmma_f32_16x16x32_bf16(false, a.v, false, b.v,
                                                   (short)0, c, false, false);
}

__device__ __forceinline__ v8f vzero8() {
    v8f x;
#pragma unroll
    for (int i = 0; i < 8; ++i) x[i] = 0.0f;
    return x;
}

__device__ __forceinline__ unsigned short f2bf(float f) {
    unsigned int u = __float_as_uint(f);
    unsigned int r = u + 0x7FFFu + ((u >> 16) & 1u);   // round-to-nearest-even
    return (unsigned short)(r >> 16);
}

// ---------------- problem constants ----------------
#define BSZ  8
#define NQL  2048
#define NKL  2048
#define QIN  256
#define QOUT 256
#define KIN  1024
#define EPSV 1e-5f

// ---------------- conversion kernels ----------------
__global__ void k_cvt_bf16(const float* __restrict__ in, unsigned short* __restrict__ out, int n) {
    int i = blockIdx.x * blockDim.x + threadIdx.x;
    if (i < n) out[i] = f2bf(in[i]);
}

// qT[b][n][c] = bf16(q[b][c][n])  via 16x16 LDS tile transpose
__global__ void k_qT_bf16(const float* __restrict__ q, unsigned short* __restrict__ qT) {
    __shared__ float tile[16][17];
    int b = blockIdx.z, n0 = blockIdx.x * 16, c0 = blockIdx.y * 16;
    int tx = threadIdx.x, ty = threadIdx.y;
    tile[ty][tx] = q[((size_t)b * QIN + (c0 + ty)) * NQL + n0 + tx];
    __syncthreads();
    qT[((size_t)b * NQL + (n0 + ty)) * QIN + c0 + tx] = f2bf(tile[tx][ty]);
}

// ---------------- generic NT bf16 GEMM ----------------
// C[m,n] = sum_k A[m,k]*B[n,k].  1 wave computes a 16x64 tile.
// TRANS==0: store bf16 row-major [M,Ntot];  TRANS==1: store bf16 transposed [Ntot,Mtot]
template <int TRANS>
__global__ __launch_bounds__(32) void k_gemm_nt(const unsigned short* __restrict__ A,
                          const unsigned short* __restrict__ Bw,
                          unsigned short* __restrict__ C,
                          int K, int lda, int ldb,
                          long long aBatch, long long cBatch,
                          int Mtot, int Ntot) {
    int lane  = threadIdx.x;
    int r15   = lane & 15;
    int akoff = (lane & 16) ? 16 : 0;
    int bkoff = (lane & 16) ? 16 : 0;
    int roff  = (lane & 16) ? 8 : 0;
    int m0 = blockIdx.x * 16;
    int n0 = blockIdx.y * 64;

    const unsigned short* Ap = A + (size_t)blockIdx.z * aBatch + (size_t)(m0 + r15) * lda;
    const unsigned short* Bp = Bw + (size_t)(n0 + r15) * ldb;

    v8f acc[4];
#pragma unroll
    for (int t = 0; t < 4; ++t) acc[t] = vzero8();

    for (int k0 = 0; k0 < K; k0 += 32) {
        Frag a;
        a.u[0] = *(const uint4*)(Ap + k0 + akoff);
        a.u[1] = *(const uint4*)(Ap + k0 + akoff + 8);
#pragma unroll
        for (int t = 0; t < 4; ++t) {
            Frag b;
            const unsigned short* bp = Bp + (size_t)t * 16 * ldb + k0 + bkoff;
            b.u[0] = *(const uint4*)(bp);
            b.u[1] = *(const uint4*)(bp + 8);
            acc[t] = wmma_bf16(a, b, acc[t]);
        }
    }

    unsigned short* Cp = C + (size_t)blockIdx.z * cBatch;
    if (TRANS == 0) {
#pragma unroll
        for (int t = 0; t < 4; ++t) {
            int col = n0 + t * 16 + r15;
#pragma unroll
            for (int v = 0; v < 8; ++v)
                Cp[(size_t)(m0 + roff + v) * Ntot + col] = f2bf(acc[t][v]);
        }
    } else {
#pragma unroll
        for (int t = 0; t < 4; ++t) {
            int col = n0 + t * 16 + r15;
            union { unsigned short h[8]; uint4 q; } pk;
#pragma unroll
            for (int v = 0; v < 8; ++v) pk.h[v] = f2bf(acc[t][v]);
            *(uint4*)(Cp + (size_t)col * Mtot + m0 + roff) = pk.q;
        }
    }
}

// ---------------- flash attention + residual ----------------
// per wave: 16 query rows. S = Xq*Xk^T/16, online softmax, O += P*Xv.
// res[b][n][c] = q[b][c][n] - O[n][c]/l[n]   (stored bf16 [n][c])
// __launch_bounds__(32, 1): single heavy wave per block, min 1 wave/EU ->
// lets the backend allocate >256 VGPRs (CDNA5 supports 1024/wave) instead of
// spilling the Xq fragments + O accumulators to scratch.
__global__ __launch_bounds__(32, 1) void k_flash(const unsigned short* __restrict__ Xq,
                        const unsigned short* __restrict__ Xk,
                        const unsigned short* __restrict__ Xvt,
                        const float* __restrict__ q,
                        unsigned short* __restrict__ res) {
    const float scale = 0.0625f;   // 1/sqrt(256)
    int lane  = threadIdx.x;
    int r15   = lane & 15;
    int akoff = (lane & 16) ? 16 : 0;
    int bkoff = (lane & 16) ? 16 : 0;
    int roff  = (lane & 16) ? 8 : 0;
    int b  = blockIdx.y;
    int n0 = blockIdx.x * 16;

    // A fragments for the full K=256 query strip, kept in registers
    const unsigned short* Aq = Xq + ((size_t)b * NQL + n0 + r15) * QOUT;
    Frag aq[8];
#pragma unroll
    for (int i = 0; i < 8; ++i) {
        aq[i].u[0] = *(const uint4*)(Aq + i * 32 + akoff);
        aq[i].u[1] = *(const uint4*)(Aq + i * 32 + akoff + 8);
    }

    v8f oacc[16];
#pragma unroll
    for (int c = 0; c < 16; ++c) oacc[c] = vzero8();
    float mi[8], li[8];
#pragma unroll
    for (int v = 0; v < 8; ++v) { mi[v] = -1e30f; li[v] = 0.0f; }

    const unsigned short* Kb = Xk + (size_t)b * NKL * QOUT;
    const unsigned short* Vb = Xvt + (size_t)b * QIN * NKL;

    __shared__ alignas(16) unsigned short Plds[16 * 32];

    for (int m0 = 0; m0 < NKL; m0 += 32) {
        v8f s0 = vzero8(), s1 = vzero8();
        const unsigned short* K0 = Kb + (size_t)(m0 + r15) * QOUT;
        const unsigned short* K1 = Kb + (size_t)(m0 + 16 + r15) * QOUT;
#pragma unroll
        for (int kc = 0; kc < 8; ++kc) {
            Frag bk0, bk1;
            bk0.u[0] = *(const uint4*)(K0 + kc * 32 + bkoff);
            bk0.u[1] = *(const uint4*)(K0 + kc * 32 + bkoff + 8);
            bk1.u[0] = *(const uint4*)(K1 + kc * 32 + bkoff);
            bk1.u[1] = *(const uint4*)(K1 + kc * 32 + bkoff + 8);
            s0 = wmma_bf16(aq[kc], bk0, s0);
            s1 = wmma_bf16(aq[kc], bk1, s1);
        }

        // online softmax over the 32 columns of this chunk
        float p0[8], p1[8];
#pragma unroll
        for (int v = 0; v < 8; ++v) { p0[v] = s0[v] * scale; p1[v] = s1[v] * scale; }
#pragma unroll
        for (int v = 0; v < 8; ++v) {
            float cm = fmaxf(p0[v], p1[v]);
            cm = fmaxf(cm, __shfl_xor(cm, 1));
            cm = fmaxf(cm, __shfl_xor(cm, 2));
            cm = fmaxf(cm, __shfl_xor(cm, 4));
            cm = fmaxf(cm, __shfl_xor(cm, 8));
            float nm   = fmaxf(mi[v], cm);
            float corr = __expf(mi[v] - nm);
            mi[v] = nm;
            li[v] *= corr;
            p0[v] = __expf(p0[v] - nm);
            p1[v] = __expf(p1[v] - nm);
            float rs = p0[v] + p1[v];
            rs += __shfl_xor(rs, 1);
            rs += __shfl_xor(rs, 2);
            rs += __shfl_xor(rs, 4);
            rs += __shfl_xor(rs, 8);
            li[v] += rs;
#pragma unroll
            for (int c = 0; c < 16; ++c) oacc[c][v] *= corr;
        }

        // P: C-layout -> A-layout via LDS (16x32 bf16)
#pragma unroll
        for (int v = 0; v < 8; ++v) {
            Plds[(roff + v) * 32 + r15]      = f2bf(p0[v]);
            Plds[(roff + v) * 32 + 16 + r15] = f2bf(p1[v]);
        }
        __syncthreads();
        Frag pf;
        pf.u[0] = *(const uint4*)&Plds[r15 * 32 + akoff];
        pf.u[1] = *(const uint4*)&Plds[r15 * 32 + akoff + 8];
        __syncthreads();

        // O accumulation over all 256 channels
#pragma unroll
        for (int c = 0; c < 16; ++c) {
            Frag bv;
            const unsigned short* vp = Vb + (size_t)(c * 16 + r15) * NKL + m0 + bkoff;
            bv.u[0] = *(const uint4*)(vp);
            bv.u[1] = *(const uint4*)(vp + 8);
            oacc[c] = wmma_bf16(pf, bv, oacc[c]);
        }
    }

    // res = q - O/l  (bf16, [n][c] row-major)
#pragma unroll
    for (int c = 0; c < 16; ++c) {
        int col = c * 16 + r15;
        const float* qp = q + ((size_t)b * QIN + col) * NQL + n0 + roff;
        unsigned short* rp = res + ((size_t)b * NQL + n0 + roff) * QIN + col;
#pragma unroll
        for (int v = 0; v < 8; ++v) {
            float o = oacc[c][v] / li[v];
            rp[(size_t)v * QIN] = f2bf(qp[v] - o);
        }
    }
}

// ---------------- Wt GEMM (transposed f32 out) + BN partial sums ----------------
__global__ __launch_bounds__(32) void k_gemm_wt(const unsigned short* __restrict__ A,   // res bf16 [b][NQ][QIN]
                          const unsigned short* __restrict__ Bw,  // Wt bf16 [QOUT][QIN]
                          float* __restrict__ Tt,                 // [b][QOUT][NQ]
                          float* __restrict__ sums) {             // [2*QOUT]
    int lane  = threadIdx.x;
    int r15   = lane & 15;
    int akoff = (lane & 16) ? 16 : 0;
    int bkoff = (lane & 16) ? 16 : 0;
    int roff  = (lane & 16) ? 8 : 0;
    int m0 = blockIdx.x * 16;
    int n0 = blockIdx.y * 64;
    int b  = blockIdx.z;

    const unsigned short* Ap = A + ((size_t)b * NQL + m0 + r15) * QIN;
    const unsigned short* Bp = Bw + (size_t)(n0 + r15) * QIN;

    v8f acc[4];
#pragma unroll
    for (int t = 0; t < 4; ++t) acc[t] = vzero8();

    for (int k0 = 0; k0 < QIN; k0 += 32) {
        Frag a;
        a.u[0] = *(const uint4*)(Ap + k0 + akoff);
        a.u[1] = *(const uint4*)(Ap + k0 + akoff + 8);
#pragma unroll
        for (int t = 0; t < 4; ++t) {
            Frag bfr;
            const unsigned short* bp = Bp + (size_t)t * 16 * QIN + k0 + bkoff;
            bfr.u[0] = *(const uint4*)(bp);
            bfr.u[1] = *(const uint4*)(bp + 8);
            acc[t] = wmma_bf16(a, bfr, acc[t]);
        }
    }

#pragma unroll
    for (int t = 0; t < 4; ++t) {
        int col = n0 + t * 16 + r15;
        float* op = Tt + ((size_t)b * QOUT + col) * NQL + m0 + roff;
        float4 lo = make_float4(acc[t][0], acc[t][1], acc[t][2], acc[t][3]);
        float4 hi = make_float4(acc[t][4], acc[t][5], acc[t][6], acc[t][7]);
        *(float4*)(op)     = lo;
        *(float4*)(op + 4) = hi;

        float s = 0.f, ss = 0.f;
#pragma unroll
        for (int v = 0; v < 8; ++v) { float x = acc[t][v]; s += x; ss += x * x; }
        s  += __shfl_xor(s, 16);
        ss += __shfl_xor(ss, 16);
        if (lane < 16) {
            atomicAdd(&sums[col], s);
            atomicAdd(&sums[QOUT + col], ss);
        }
    }
}

// ---------------- BN stats / epilogue ----------------
__global__ void k_zero(float* __restrict__ p, int n) {
    int i = blockIdx.x * blockDim.x + threadIdx.x;
    if (i < n) p[i] = 0.0f;
}

__global__ void k_bnstats(const float* __restrict__ sums,
                          const float* __restrict__ gamma,
                          const float* __restrict__ beta,
                          float* __restrict__ sc) {
    int o = threadIdx.x;  // 256
    float n    = (float)(BSZ * NQL);
    float mean = sums[o] / n;
    float var  = sums[QOUT + o] / n - mean * mean;
    float g    = gamma[o] * rsqrtf(var + EPSV);
    sc[o]        = g;
    sc[QOUT + o] = beta[o] - mean * g;
}

__global__ void k_final(const float* __restrict__ Tt,  // [b][c][n], same layout as q/out
                        const float* __restrict__ sc,
                        const float* __restrict__ q,
                        float* __restrict__ out) {
    size_t i = (size_t)blockIdx.x * blockDim.x + threadIdx.x;
    size_t total = (size_t)BSZ * QIN * NQL;
    if (i >= total) return;
    int c = (int)((i / NQL) % QIN);
    float v = fmaf(Tt[i], sc[c], sc[QOUT + c]);
    out[i] = fmaxf(v, 0.0f) + q[i];
}

// ---------------- host launch ----------------
extern "C" void kernel_launch(void* const* d_in, const int* in_sizes, int n_in,
                              void* d_out, int out_size, void* d_ws, size_t ws_size,
                              hipStream_t stream) {
    const float* q   = (const float*)d_in[0];  // [8,256,2048]
    const float* kv  = (const float*)d_in[1];  // [8,2048,1024]
    const float* Wq  = (const float*)d_in[2];  // [256,256]
    const float* Wk  = (const float*)d_in[3];  // [256,1024]
    const float* Wv  = (const float*)d_in[4];  // [256,1024]
    const float* Wt  = (const float*)d_in[5];  // [256,256]
    // d_in[6] = bt  : cancels exactly under training-mode BN -> unused
    const float* gamma = (const float*)d_in[7];
    const float* beta  = (const float*)d_in[8];
    float* out = (float*)d_out;

    char* w = (char*)d_ws;
    size_t off = 0;
    unsigned short* qT   = (unsigned short*)(w + off); off += (size_t)BSZ * NQL * QIN * 2;   // 8 MB
    unsigned short* kvbf = (unsigned short*)(w + off); off += (size_t)BSZ * NKL * KIN * 2;   // 32 MB
    unsigned short* wqbf = (unsigned short*)(w + off); off += (size_t)QOUT * QIN * 2;
    unsigned short* wkbf = (unsigned short*)(w + off); off += (size_t)QOUT * KIN * 2;
    unsigned short* wvbf = (unsigned short*)(w + off); off += (size_t)QIN * KIN * 2;
    unsigned short* wtbf = (unsigned short*)(w + off); off += (size_t)QIN * QIN * 2;
    unsigned short* Xq   = (unsigned short*)(w + off); off += (size_t)BSZ * NQL * QOUT * 2;  // 8 MB
    unsigned short* Xk   = (unsigned short*)(w + off); off += (size_t)BSZ * NKL * QOUT * 2;  // 8 MB
    unsigned short* Xvt  = (unsigned short*)(w + off); off += (size_t)BSZ * QIN * NKL * 2;   // 8 MB
    unsigned short* res  = (unsigned short*)(w + off); off += (size_t)BSZ * NQL * QIN * 2;   // 8 MB
    float* Tt   = (float*)(w + off); off += (size_t)BSZ * QOUT * NQL * 4;                    // 16 MB
    float* sums = (float*)(w + off); off += 2 * QOUT * 4;
    float* sc   = (float*)(w + off); off += 2 * QOUT * 4;
    (void)ws_size; (void)n_in; (void)in_sizes; (void)out_size;

    // 1) conversions
    {
        int n;
        n = QOUT * QIN;  k_cvt_bf16<<<(n + 255) / 256, 256, 0, stream>>>(Wq, wqbf, n);
        n = QOUT * KIN;  k_cvt_bf16<<<(n + 255) / 256, 256, 0, stream>>>(Wk, wkbf, n);
        n = QIN  * KIN;  k_cvt_bf16<<<(n + 255) / 256, 256, 0, stream>>>(Wv, wvbf, n);
        n = QIN  * QIN;  k_cvt_bf16<<<(n + 255) / 256, 256, 0, stream>>>(Wt, wtbf, n);
        int nk = BSZ * NKL * KIN;
        k_cvt_bf16<<<(nk + 255) / 256, 256, 0, stream>>>(kv, kvbf, nk);
        k_qT_bf16<<<dim3(NQL / 16, QIN / 16, BSZ), dim3(16, 16), 0, stream>>>(q, qT);
    }
    // 2) projections
    k_gemm_nt<0><<<dim3(NQL / 16, QOUT / 64, BSZ), 32, 0, stream>>>(
        qT, wqbf, Xq, QIN, QIN, QIN,
        (long long)NQL * QIN, (long long)NQL * QOUT, NQL, QOUT);
    k_gemm_nt<0><<<dim3(NKL / 16, QOUT / 64, BSZ), 32, 0, stream>>>(
        kvbf, wkbf, Xk, KIN, KIN, KIN,
        (long long)NKL * KIN, (long long)NKL * QOUT, NKL, QOUT);
    k_gemm_nt<1><<<dim3(NKL / 16, QIN / 64, BSZ), 32, 0, stream>>>(
        kvbf, wvbf, Xvt, KIN, KIN, KIN,
        (long long)NKL * KIN, (long long)QIN * NKL, NKL, QIN);
    // 3) BN accumulators
    k_zero<<<1, 2 * QOUT, 0, stream>>>(sums, 2 * QOUT);
    // 4) fused attention -> res
    k_flash<<<dim3(NQL / 16, BSZ), 32, 0, stream>>>(Xq, Xk, Xvt, q, res);
    // 5) Wt GEMM + BN partial sums
    k_gemm_wt<<<dim3(NQL / 16, QOUT / 64, BSZ), 32, 0, stream>>>(res, wtbf, Tt, sums);
    // 6) BN stats -> scale/shift
    k_bnstats<<<1, QOUT, 0, stream>>>(sums, gamma, beta, sc);
    // 7) epilogue
    {
        size_t total = (size_t)BSZ * QIN * NQL;
        k_final<<<(unsigned)((total + 255) / 256), 256, 0, stream>>>(Tt, sc, q, out);
    }
}
